// FourierParametrization_25168508355384
// MI455X (gfx1250) — matrix operand order repaired and checked by reading, not compiled, not run on hardware
//
#include <hip/hip_runtime.h>
#include <stdint.h>

// ============================================================================
// out = original_weights + (α/n) · Re( ifft2(sparse F) )
//
// Sparse spectrum => closed form:
//   w[m,p] = S · Σ_j c_j cos(2π(m·k_j + p·l_j)/4096),  S = 1/(1024·4096·4096)
// cos(a+b) = cos a cos b − sin a sin b  =>  single f16 GEMM, K = 2·1024:
//   A'[4096×2048] : [cos(2π m k_j/4096) | sin(2π m k_j/4096)]
//   Bt[4096×2048] : [c_j cos(2π p l_j/4096) | −c_j sin(2π p l_j/4096)]   (K-major)
//   w = S · (A' · Btᵀ)    — mapped onto v_wmma_f32_16x16x32_f16
//
// Data movement: double-buffered GLOBAL_LOAD_ASYNC_TO_LDS_B128 (ASYNCcnt),
// no VGPR staging; compute overlaps the async engine's transfer.
// ============================================================================

typedef _Float16 v16h __attribute__((ext_vector_type(16)));
typedef _Float16 v8h  __attribute__((ext_vector_type(8)));
typedef float    v8f  __attribute__((ext_vector_type(8)));

#define FOUT 4096
#define FIN  4096
#define NC   1024
#define KDIM 2048                      // cos block | sin block
#define TWO_PI_OVER_4096 1.5339807878856412e-3f
#define OUT_SCALE 5.8207660913467407e-11f   // 1/(1024*4096*4096)

#define BLK   128                      // block tile (M and N)
#define BK    32                       // K step == one WMMA K
#define LDSS  40                       // padded LDS row stride in halves (80B, 16B-aligned)

// ---------------------------------------------------------------------------
// Table builders (trivial cost: 8.4M sincos total vs 17.2G for brute force)
// ---------------------------------------------------------------------------
__global__ __launch_bounds__(256) void build_a_tab(const int* __restrict__ E,
                                                   _Float16* __restrict__ A) {
    const int j = blockIdx.x * blockDim.x + threadIdx.x;   // coeff 0..1023
    const int m = blockIdx.y;                              // row   0..4095
    const int k = E[j];                                    // E[0][j]
    const float th = (float)((m * k) & 4095) * TWO_PI_OVER_4096;
    float s, c;
    __sincosf(th, &s, &c);
    A[(size_t)m * KDIM + j]      = (_Float16)c;
    A[(size_t)m * KDIM + NC + j] = (_Float16)s;
}

__global__ __launch_bounds__(256) void build_b_tab(const int* __restrict__ E,
                                                   const float* __restrict__ cv,
                                                   _Float16* __restrict__ Bt) {
    const int j = blockIdx.x * blockDim.x + threadIdx.x;   // coeff 0..1023
    const int p = blockIdx.y;                              // col   0..4095
    const int l = E[NC + j];                               // E[1][j]
    const float cj = cv[j];
    const float th = (float)((p * l) & 4095) * TWO_PI_OVER_4096;
    float s, c;
    __sincosf(th, &s, &c);
    Bt[(size_t)p * KDIM + j]      = (_Float16)( cj * c);
    Bt[(size_t)p * KDIM + NC + j] = (_Float16)(-cj * s);   // minus folded in here
}

// ---------------------------------------------------------------------------
// Async global -> LDS, 16B per lane, tracked by ASYNCcnt (no VGPR staging).
// ldsoff: wave-relative LDS byte address (low 32 bits of the generic pointer,
// per the flat-aperture mapping).  gaddr: 64-bit global byte address.
// ---------------------------------------------------------------------------
__device__ __forceinline__ void async_copy_b128(uint32_t ldsoff, const void* gaddr) {
    asm volatile("global_load_async_to_lds_b128 %0, %1, off"
                 :: "v"(ldsoff), "v"((uint64_t)(uintptr_t)gaddr)
                 : "memory");
}
__device__ __forceinline__ void wait_asynccnt0() {
    asm volatile("s_wait_asynccnt 0x0" ::: "memory");
}

// ---------------------------------------------------------------------------
// WMMA GEMM: out[m][p] = orig[m][p] + S * Σ_k A'[m][k]·Bt[p][k]
//   grid 32×32, 256 threads = 8 waves (2×4), wave tile 64×32 = 4×2 WMMA accs
//   double-buffered LDS tiles filled by the async engine
// ---------------------------------------------------------------------------
__global__ __launch_bounds__(256) void fourier_wmma_gemm(
        const _Float16* __restrict__ A, const _Float16* __restrict__ Bt,
        const float* __restrict__ orig, float* __restrict__ out) {
    __shared__ _Float16 la[2][BLK * LDSS];   // A tiles: 128 rows × 32 k (padded)
    __shared__ _Float16 lb[2][BLK * LDSS];   // B tiles: 128 cols × 32 k (padded)

    const int tid  = threadIdx.x;
    const int lane = tid & 31;
    const int wave = tid >> 5;
    const int wm = wave >> 2;            // wave row  0..1  (64 rows each)
    const int wn = wave &  3;            // wave col  0..3  (32 cols each)
    const int m0 = blockIdx.y * BLK;
    const int n0 = blockIdx.x * BLK;

    const int nloc  = lane & 15;         // fragment row/col within 16
    const int half  = lane >> 4;         // lane group 0/1
    const int kaoff = half * 8;          // A frag: K ∈ {ka..+7} ∪ {ka+16..+23}
    const int kboff = half * 8;          // B frag: K ∈ {kb..+7} ∪ {kb+16..+23}

    // ---- per-thread async-copy slots: 2 chunks per tile, 16B each ----------
    const int row0 = tid >> 2;           // tile row 0..63
    const int row1 = row0 + 64;          // tile row 64..127
    const int cq   = (tid & 3) * 8;      // 8-half chunk within a 32-half row
    const _Float16* gA0 = A  + (size_t)(m0 + row0) * KDIM + cq;
    const _Float16* gA1 = A  + (size_t)(m0 + row1) * KDIM + cq;
    const _Float16* gB0 = Bt + (size_t)(n0 + row0) * KDIM + cq;
    const _Float16* gB1 = Bt + (size_t)(n0 + row1) * KDIM + cq;

    #define ISSUE_TILE(k0, s)                                                   \
        do {                                                                    \
            async_copy_b128((uint32_t)(uintptr_t)&la[s][row0 * LDSS + cq], gA0 + (k0)); \
            async_copy_b128((uint32_t)(uintptr_t)&la[s][row1 * LDSS + cq], gA1 + (k0)); \
            async_copy_b128((uint32_t)(uintptr_t)&lb[s][row0 * LDSS + cq], gB0 + (k0)); \
            async_copy_b128((uint32_t)(uintptr_t)&lb[s][row1 * LDSS + cq], gB1 + (k0)); \
        } while (0)

    v8f acc[4][2];
    const v8f vzero = {0.f, 0.f, 0.f, 0.f, 0.f, 0.f, 0.f, 0.f};
#pragma unroll
    for (int mi = 0; mi < 4; ++mi)
#pragma unroll
        for (int ni = 0; ni < 2; ++ni) acc[mi][ni] = vzero;

    // ---- software pipeline: prefetch tile 0, then overlap next-tile async
    //      transfer with current-tile WMMA work -----------------------------
    ISSUE_TILE(0, 0);
    int sel = 0;
    for (int k0 = 0; k0 < KDIM; k0 += BK) {
        wait_asynccnt0();      // this thread's async writes for tile(sel) done
        __syncthreads();       // ...and everyone's; also: all reads of the
                               // other buffer (prev iter compute) are done,
                               // so it is safe to overwrite it below.
        if (k0 + BK < KDIM) ISSUE_TILE(k0 + BK, sel ^ 1);

        // ---- fragment loads per ISA §7.12.2 layouts ------------------------
        union { v16h v; v8h h[2]; } af[4], bf[2];
#pragma unroll
        for (int ni = 0; ni < 2; ++ni) {
            const int n = wn * 32 + ni * 16 + nloc;
            bf[ni].h[0] = *(const v8h*)&lb[sel][n * LDSS + kboff];
            bf[ni].h[1] = *(const v8h*)&lb[sel][n * LDSS + kboff + 16];
        }
#pragma unroll
        for (int mi = 0; mi < 4; ++mi) {
            const int m = wm * 64 + mi * 16 + nloc;
            af[mi].h[0] = *(const v8h*)&la[sel][m * LDSS + kaoff];
            af[mi].h[1] = *(const v8h*)&la[sel][m * LDSS + kaoff + 16];
        }

        // ---- 8 WMMAs per wave per K-step ----------------------------------
#pragma unroll
        for (int mi = 0; mi < 4; ++mi)
#pragma unroll
            for (int ni = 0; ni < 2; ++ni)
                acc[mi][ni] = __builtin_amdgcn_wmma_f32_16x16x32_f16(
                    false, af[mi].v, false, bf[ni].v,
                    (short)0, acc[mi][ni], false, false);
        sel ^= 1;
    }
    #undef ISSUE_TILE

    // ---- epilogue: fused residual add, C/D layout -> coalesced f32 stores --
#pragma unroll
    for (int mi = 0; mi < 4; ++mi)
#pragma unroll
        for (int ni = 0; ni < 2; ++ni) {
            const int col  = n0 + wn * 32 + ni * 16 + nloc;
            const int mrow = m0 + wm * 64 + mi * 16 + half * 8;
#pragma unroll
            for (int r = 0; r < 8; ++r) {
                const size_t idx = (size_t)(mrow + r) * FIN + col;
                out[idx] = orig[idx] + OUT_SCALE * acc[mi][ni][r];
            }
        }
}

// ---------------------------------------------------------------------------
// Fallback (no workspace): direct cosine sum. Correct, slow; only used if the
// harness workspace is smaller than the 32MB of f16 tables we need.
// ---------------------------------------------------------------------------
__global__ __launch_bounds__(256) void fourier_direct(
        const float* __restrict__ orig, const float* __restrict__ cv,
        const int* __restrict__ E, float* __restrict__ out) {
    __shared__ int   sk[NC];
    __shared__ int   sl[NC];
    __shared__ float sc[NC];
    for (int j = threadIdx.x; j < NC; j += blockDim.x) {
        sk[j] = E[j]; sl[j] = E[NC + j]; sc[j] = cv[j];
    }
    __syncthreads();
    const size_t gid = (size_t)blockIdx.x * blockDim.x + threadIdx.x;
    const int m = (int)(gid >> 12);       // /4096
    const int p = (int)(gid & 4095);
    float sum = 0.f;
    for (int j = 0; j < NC; ++j) {
        const int ph = (m * sk[j] + p * sl[j]) & 4095;
        sum += sc[j] * __cosf((float)ph * TWO_PI_OVER_4096);
    }
    out[gid] = orig[gid] + OUT_SCALE * sum;
}

extern "C" void kernel_launch(void* const* d_in, const int* in_sizes, int n_in,
                              void* d_out, int out_size, void* d_ws, size_t ws_size,
                              hipStream_t stream) {
    const float* orig = (const float*)d_in[0];   // [4096*4096] f32
    const float* cv   = (const float*)d_in[1];   // [1024] f32
    const int*   E    = (const int*)d_in[2];     // [2*1024] i32
    float* out = (float*)d_out;

    const size_t tabBytes = (size_t)FOUT * KDIM * sizeof(_Float16);  // 16 MB each
    if (ws_size >= 2 * tabBytes) {
        _Float16* A  = (_Float16*)d_ws;
        _Float16* Bt = (_Float16*)((char*)d_ws + tabBytes);
        build_a_tab<<<dim3(NC / 256, FOUT), 256, 0, stream>>>(E, A);
        build_b_tab<<<dim3(NC / 256, FIN),  256, 0, stream>>>(E, cv, Bt);
        fourier_wmma_gemm<<<dim3(FIN / BLK, FOUT / BLK), 256, 0, stream>>>(A, Bt, orig, out);
    } else {
        fourier_direct<<<(FOUT * FIN) / 256, 256, 0, stream>>>(orig, cv, E, out);
    }
}
